// FastNystromAttention_8108898255624
// MI455X (gfx1250) — matrix operand order repaired and checked by reading, not compiled, not run on hardware
//
#include <hip/hip_runtime.h>

// ---------------------------------------------------------------------------
// FastNystromAttention on MI455X (gfx1250, wave32, WMMA bf16 16x16x32)
// ---------------------------------------------------------------------------
namespace {
constexpr int kB = 2, kN = 2048, kDM = 1024, kH = 16, kM = 64, kHD = 64;
constexpr int kMAXLEN = 2048;
constexpr int kBH = kB * kH;            // 32
constexpr float kScale = 0.125f;        // 1/sqrt(64)
}

typedef __attribute__((ext_vector_type(16))) __bf16 v16bf;
typedef __attribute__((ext_vector_type(8)))  float  v8f;
typedef __attribute__((ext_vector_type(4)))  int    v4i_t;

union Frag {
  v16bf v;
  uint4 u[2];
};

// ---- gfx1250 async global->LDS copy (guarded; falls back to sync copy) ----
#if defined(__has_builtin)
#if __has_builtin(__builtin_amdgcn_global_load_async_to_lds_b128) && \
    __has_builtin(__builtin_amdgcn_s_wait_asynccnt)
#define ASYNC_LDS 1
#endif
#endif
#ifndef ASYNC_LDS
#define ASYNC_LDS 0
#endif

__device__ __forceinline__ void copy16_g2l(const unsigned short* g, unsigned short* l) {
#if ASYNC_LDS
  __builtin_amdgcn_global_load_async_to_lds_b128(
      (__attribute__((address_space(1))) v4i_t*)(v4i_t*)const_cast<unsigned short*>(g),
      (__attribute__((address_space(3))) v4i_t*)(v4i_t*)l,
      0, 0);
#else
  *(uint4*)l = *(const uint4*)g;
#endif
}
__device__ __forceinline__ void async_wait() {
#if ASYNC_LDS
  __builtin_amdgcn_s_wait_asynccnt(0);
#endif
}

__device__ __forceinline__ unsigned short f32_to_bf16(float f) {
  unsigned int u = __float_as_uint(f);
  u += 0x7fffu + ((u >> 16) & 1u);      // round-to-nearest-even
  return (unsigned short)(u >> 16);
}
__device__ __forceinline__ float bf16_to_f32(unsigned short h) {
  return __uint_as_float(((unsigned int)h) << 16);
}
__device__ __forceinline__ v8f v8f_zero() {
  v8f v;
#pragma unroll
  for (int i = 0; i < 8; ++i) v[i] = 0.f;
  return v;
}

// A-matrix 16x32 bf16 fragment from row-major [16][>=32] (stride ld elements).
__device__ __forceinline__ v16bf load_a_frag(const unsigned short* base, int ld, int lane) {
  int m = lane & 15;
  int o = (lane >> 4) * 8;
  const unsigned short* p = base + (size_t)m * ld;
  Frag f;
  f.u[0] = *(const uint4*)(p + o);
  f.u[1] = *(const uint4*)(p + o + 16);
  return f.v;
}

// B-matrix 32x16 bf16 fragment from TRANSPOSED storage [n][k] (stride ld).
__device__ __forceinline__ v16bf load_b_frag(const unsigned short* baseT, int ld, int lane) {
  int n = lane & 15;
  int ko = (lane >> 4) * 8;
  const unsigned short* p = baseT + (size_t)n * ld + ko;
  Frag f;
  f.u[0] = *(const uint4*)(p);
  f.u[1] = *(const uint4*)(p + 16);
  return f.v;
}

__device__ __forceinline__ v8f wmma_bf16(v16bf a, v16bf b, v8f c) {
  return __builtin_amdgcn_wmma_f32_16x16x32_bf16(false, a, false, b, (short)0, c, false, false);
}

// ---------------------------------------------------------------------------
// f32 -> bf16 conversion (vectorized x4) — used for activations (x)
// ---------------------------------------------------------------------------
__global__ void cvt_kernel(const float* __restrict__ s, unsigned short* __restrict__ d, int n4) {
  int i = blockIdx.x * blockDim.x + threadIdx.x;
  if (i >= n4) return;
  float4 f = ((const float4*)s)[i];
  ushort4 o;
  o.x = f32_to_bf16(f.x); o.y = f32_to_bf16(f.y);
  o.z = f32_to_bf16(f.z); o.w = f32_to_bf16(f.w);
  ((ushort4*)d)[i] = o;
}

// ---------------------------------------------------------------------------
// Weight transpose+convert: wt[n][k] (bf16) = w[k][n] (f32). 32x32 LDS tiles.
// ---------------------------------------------------------------------------
__global__ __launch_bounds__(256)
void transpose_cvt(const float* __restrict__ w, unsigned short* __restrict__ wt,
                   int K, int Nn) {
  __shared__ float tile[32][33];
  int tx = threadIdx.x & 31, ty = threadIdx.x >> 5;   // 32 x 8
  int n0 = blockIdx.x * 32, k0 = blockIdx.y * 32;
#pragma unroll
  for (int i = 0; i < 4; ++i)
    tile[ty + i * 8][tx] = w[(size_t)(k0 + ty + i * 8) * Nn + n0 + tx];
  __syncthreads();
#pragma unroll
  for (int i = 0; i < 4; ++i)
    wt[(size_t)(n0 + ty + i * 8) * K + k0 + tx] = f32_to_bf16(tile[tx][ty + i * 8]);
}

// ---------------------------------------------------------------------------
// bf16 WMMA GEMM: C[M,N] = A[M,K] @ Bt[N,K]^T + bias[N]   (f32 out)
// Block 256 thr = 8 waves, tile 128x128, K-step 32, double-buffered LDS,
// async global->LDS staging when available.
// ---------------------------------------------------------------------------
__global__ __launch_bounds__(256)
void gemm_bf16(const unsigned short* __restrict__ Amat,
               const unsigned short* __restrict__ Bt,
               const float* __restrict__ bias, float* __restrict__ Cmat,
               int Mdim, int Ndim, int Kdim) {
  __shared__ __align__(16) unsigned short a_lds[2][128][32];
  __shared__ __align__(16) unsigned short b_lds[2][128][32];   // [n][k]
  int tid = threadIdx.x, lane = tid & 31, w = tid >> 5;
  int wm = w >> 2, wn = w & 3;                                 // 2x4 wave grid
  int m0 = blockIdx.y * 128, n0 = blockIdx.x * 128;

  // per-thread staging coords: 2 x 16B per tile
  int sidx0 = tid * 16, sidx1 = tid * 16 + 8;
  int sr0 = sidx0 >> 5, sc0 = sidx0 & 31;
  int sr1 = sidx1 >> 5, sc1 = sidx1 & 31;
  const unsigned short* ga0 = &Amat[(size_t)(m0 + sr0) * Kdim + sc0];
  const unsigned short* ga1 = &Amat[(size_t)(m0 + sr1) * Kdim + sc1];
  const unsigned short* gb0 = &Bt[(size_t)(n0 + sr0) * Kdim + sc0];
  const unsigned short* gb1 = &Bt[(size_t)(n0 + sr1) * Kdim + sc1];

  v8f acc[4][2];
#pragma unroll
  for (int mt = 0; mt < 4; ++mt)
#pragma unroll
    for (int nt = 0; nt < 2; ++nt) acc[mt][nt] = v8f_zero();

  int nk = Kdim >> 5;
  // prologue: stage tile 0 into buffer 0
  copy16_g2l(ga0, &a_lds[0][sr0][sc0]);
  copy16_g2l(ga1, &a_lds[0][sr1][sc1]);
  copy16_g2l(gb0, &b_lds[0][sr0][sc0]);
  copy16_g2l(gb1, &b_lds[0][sr1][sc1]);

  for (int kt = 0; kt < nk; ++kt) {
    async_wait();
    __syncthreads();                       // tile kt visible in buf kt&1
    if (kt + 1 < nk) {
      int nb = (kt + 1) & 1, ko = (kt + 1) * 32;
      copy16_g2l(ga0 + ko, &a_lds[nb][sr0][sc0]);
      copy16_g2l(ga1 + ko, &a_lds[nb][sr1][sc1]);
      copy16_g2l(gb0 + ko, &b_lds[nb][sr0][sc0]);
      copy16_g2l(gb1 + ko, &b_lds[nb][sr1][sc1]);
    }
    int buf = kt & 1;
    v16bf af[4];
#pragma unroll
    for (int mt = 0; mt < 4; ++mt) af[mt] = load_a_frag(&a_lds[buf][wm * 64 + mt * 16][0], 32, lane);
    v16bf bfr[2];
#pragma unroll
    for (int nt = 0; nt < 2; ++nt) bfr[nt] = load_b_frag(&b_lds[buf][wn * 32 + nt * 16][0], 32, lane);
#pragma unroll
    for (int mt = 0; mt < 4; ++mt)
#pragma unroll
      for (int nt = 0; nt < 2; ++nt)
        acc[mt][nt] = wmma_bf16(af[mt], bfr[nt], acc[mt][nt]);
  }

  int hi = lane >> 4, ln = lane & 15;
  float* cp = Cmat + (size_t)(m0 + wm * 64) * Ndim + (n0 + wn * 32);
#pragma unroll
  for (int nt = 0; nt < 2; ++nt) {
    float bv = bias[n0 + wn * 32 + nt * 16 + ln];
#pragma unroll
    for (int mt = 0; mt < 4; ++mt)
#pragma unroll
      for (int r = 0; r < 8; ++r)
        cp[(mt * 16 + r + 8 * hi) * Ndim + nt * 16 + ln] = acc[mt][nt][r] + bv;
  }
}

// ---------------------------------------------------------------------------
// Repack qkv f32 (B,N,3,H,HD) -> q/k/v bf16 (B,H,N,HD)
// ---------------------------------------------------------------------------
__global__ void repack_qkv(const float* __restrict__ qkv,
                           unsigned short* __restrict__ q,
                           unsigned short* __restrict__ k,
                           unsigned short* __restrict__ v) {
  int i = blockIdx.x * blockDim.x + threadIdx.x;
  if (i >= kB * kN * 3 * kDM) return;
  int f  = i & 63;
  int c2 = i >> 6;
  int h  = c2 & 15;
  int c3 = c2 >> 4;                // (b*N+n)*3 + s
  int s  = c3 % 3;
  int row = c3 / 3;
  int n = row & (kN - 1);
  int b = row >> 11;
  unsigned short bv = f32_to_bf16(qkv[i]);
  size_t dst = (((size_t)(b * kH + h)) * kN + n) * 64 + f;
  if (s == 0) q[dst] = bv; else if (s == 1) k[dst] = bv; else v[dst] = bv;
}

// gather landmark keys: kl (B,H,M,HD) bf16
__global__ void gather_kl(const unsigned short* __restrict__ k,
                          const int* __restrict__ lidx,
                          unsigned short* __restrict__ kl) {
  int i = blockIdx.x * blockDim.x + threadIdx.x;
  if (i >= kBH * kM * kHD) return;
  int f = i & 63, j = (i >> 6) & 63, bh = i >> 12;
  kl[i] = k[((size_t)bh * kN + lidx[j]) * 64 + f];
}

// ---------------------------------------------------------------------------
// rel_bias[bh,i,j] = q[bh,i,:] . emb[i-j+MAXLEN-1,:]   (j = 0..63)
// ---------------------------------------------------------------------------
__global__ __launch_bounds__(256)
void rel_bias_kernel(const unsigned short* __restrict__ qbf,
                     const float* __restrict__ emb, float* __restrict__ rb) {
  __shared__ float qrow[4][64];
  int blk = blockIdx.x;
  int i0 = (blk & (kN / 4 - 1)) * 4;
  int bh = blk / (kN / 4);
  int tid = threadIdx.x;
  {
    int r = tid >> 6, f = tid & 63;
    qrow[r][f] = bf16_to_f32(qbf[((size_t)bh * kN + i0 + r) * 64 + f]);
  }
  __syncthreads();
  int r = tid >> 6, j = tid & 63;
  int i = i0 + r;
  const float4* e4 = (const float4*)&emb[(size_t)(i - j + kMAXLEN - 1) * 64];
  float s = 0.f;
#pragma unroll
  for (int f4 = 0; f4 < 16; ++f4) {
    float4 ev = e4[f4];
    s += qrow[r][f4 * 4 + 0] * ev.x + qrow[r][f4 * 4 + 1] * ev.y +
         qrow[r][f4 * 4 + 2] * ev.z + qrow[r][f4 * 4 + 3] * ev.w;
  }
  rb[((size_t)bh * kN + i) * 64 + j] = s;
}

// ---------------------------------------------------------------------------
// Newton-Schulz pinv of kernel2 = scale*kl@kl^T + 1e-6 I  (64x64 per bh)
// ---------------------------------------------------------------------------
__global__ __launch_bounds__(256)
void pinv_kernel(const unsigned short* __restrict__ kl, float* __restrict__ k2inv) {
  __shared__ float G[64 * 64];
  __shared__ float X[64 * 64];
  __shared__ float Zm[64 * 64];
  __shared__ float rs[64];
  __shared__ float inv_n2;
  int bh = blockIdx.x, tid = threadIdx.x;
  const unsigned short* klp = kl + (size_t)bh * 4096;
  for (int e = tid; e < 4096; e += 256) X[e] = bf16_to_f32(klp[e]);  // stage kl in X
  __syncthreads();
  for (int e = tid; e < 4096; e += 256) {
    int i = e >> 6, j = e & 63;
    float s = 0.f;
    for (int f = 0; f < 64; ++f) s += X[i * 64 + f] * X[j * 64 + f];
    s *= kScale;
    if (i == j) s += 1e-6f;
    G[e] = s;
  }
  __syncthreads();
  if (tid < 64) {
    float s = 0.f;
    for (int j = 0; j < 64; ++j) s += fabsf(G[tid * 64 + j]);
    rs[tid] = s;
  }
  __syncthreads();
  if (tid == 0) {
    float mx = 0.f;
    for (int j = 0; j < 64; ++j) mx = fmaxf(mx, rs[j]);
    inv_n2 = 1.0f / (mx * mx);      // symmetric: ||G||_1 == ||G||_inf
  }
  __syncthreads();
  float s0 = inv_n2;
  for (int e = tid; e < 4096; e += 256) X[e] = G[e] * s0;
  __syncthreads();
  for (int it = 0; it < 25; ++it) {
    for (int e = tid; e < 4096; e += 256) {
      int i = e >> 6, j = e & 63;
      float s = 0.f;
      for (int f = 0; f < 64; ++f) s += G[i * 64 + f] * X[f * 64 + j];
      Zm[e] = s;
    }
    __syncthreads();
    float regs[16];
    int p = 0;
    for (int e = tid; e < 4096; e += 256, ++p) {
      int i = e >> 6, j = e & 63;
      float s = 0.f;
      for (int f = 0; f < 64; ++f) s += X[i * 64 + f] * Zm[f * 64 + j];
      regs[p] = s;
    }
    __syncthreads();
    p = 0;
    for (int e = tid; e < 4096; e += 256, ++p) X[e] = 2.0f * X[e] - regs[p];
    __syncthreads();
  }
  for (int e = tid; e < 4096; e += 256) k2inv[(size_t)bh * 4096 + e] = X[e];
}

// ---------------------------------------------------------------------------
// form_eff: mode 0: keff = (k @ kl^T)*scale                       -> bf16
//           mode 1: qeff = ((q @ kl^T)*scale + rel_bias) @ k2inv  -> bf16
// Block 128 thr = 4 waves, 64 rows / block, grid = BH * N/64.
// ---------------------------------------------------------------------------
__global__ __launch_bounds__(128)
void form_eff(const unsigned short* __restrict__ src,
              const unsigned short* __restrict__ kl,
              const float* __restrict__ rb,
              const float* __restrict__ k2inv,
              unsigned short* __restrict__ dst, int mode) {
  __shared__ __align__(16) unsigned short kl_lds[64 * 64];   // [j][f]
  __shared__ __align__(16) unsigned short k2_lds[64 * 64];   // symmetric -> [m][j]
  __shared__ __align__(16) unsigned short t_lds[4][16 * 64];
  int tid = threadIdx.x, lane = tid & 31, w = tid >> 5;
  int blk = blockIdx.x;
  int tile = blk & 31, bh = blk >> 5;
  int i0 = tile * 64 + w * 16;

  const unsigned short* klp = kl + (size_t)bh * 4096;
  for (int e8 = tid; e8 < 512; e8 += 128)
    copy16_g2l(klp + e8 * 8, kl_lds + e8 * 8);
  if (mode) {
    const float* kp = k2inv + (size_t)bh * 4096;
    for (int e = tid; e < 4096; e += 128) k2_lds[e] = f32_to_bf16(kp[e]);
  }
  async_wait();
  __syncthreads();

  const unsigned short* arow = src + (((size_t)bh) * kN + i0) * 64;
  v16bf af[2];
  af[0] = load_a_frag(arow, 64, lane);
  af[1] = load_a_frag(arow + 32, 64, lane);

  v8f acc[4];
#pragma unroll
  for (int jt = 0; jt < 4; ++jt) acc[jt] = v8f_zero();
#pragma unroll
  for (int jt = 0; jt < 4; ++jt)
#pragma unroll
    for (int kf = 0; kf < 2; ++kf)
      acc[jt] = wmma_bf16(af[kf], load_b_frag(&kl_lds[(jt * 16) * 64 + kf * 32], 64, lane), acc[jt]);

  int hi = lane >> 4, ln = lane & 15;
  unsigned short* dp = dst + ((size_t)bh * kN + i0) * 64;
  const float* rp = rb + ((size_t)bh * kN + i0) * 64;
  if (mode == 0) {
#pragma unroll
    for (int jt = 0; jt < 4; ++jt)
#pragma unroll
      for (int r = 0; r < 8; ++r)
        dp[(r + 8 * hi) * 64 + jt * 16 + ln] = f32_to_bf16(acc[jt][r] * kScale);
  } else {
#pragma unroll
    for (int jt = 0; jt < 4; ++jt)
#pragma unroll
      for (int r = 0; r < 8; ++r) {
        int j = jt * 16 + ln, rr = r + 8 * hi;
        float tv = acc[jt][r] * kScale + rp[rr * 64 + j];
        t_lds[w][rr * 64 + j] = f32_to_bf16(tv);
      }
    __syncthreads();
    v16bf at[2];
    at[0] = load_a_frag(&t_lds[w][0], 64, lane);
    at[1] = load_a_frag(&t_lds[w][32], 64, lane);
    v8f acc2[4];
#pragma unroll
    for (int nt = 0; nt < 4; ++nt) acc2[nt] = v8f_zero();
#pragma unroll
    for (int nt = 0; nt < 4; ++nt)
#pragma unroll
      for (int jf = 0; jf < 2; ++jf)
        acc2[nt] = wmma_bf16(at[jf], load_b_frag(&k2_lds[(nt * 16) * 64 + jf * 32], 64, lane), acc2[nt]);
#pragma unroll
    for (int nt = 0; nt < 4; ++nt)
#pragma unroll
      for (int r = 0; r < 8; ++r)
        dp[(r + 8 * hi) * 64 + nt * 16 + ln] = f32_to_bf16(acc2[nt][r]);
  }
}

// ---------------------------------------------------------------------------
// Flash attention over effective features (head-dim 64):
//   out = softmax(qeff @ keff^T) @ v ; keys streamed in chunks of 32.
// Block 128 thr = 4 waves (16 query rows each). Writes bf16 (B,N,DM).
// ---------------------------------------------------------------------------
__global__ __launch_bounds__(128)
void flash_kernel(const unsigned short* __restrict__ qe,
                  const unsigned short* __restrict__ ke,
                  const unsigned short* __restrict__ vb,
                  unsigned short* __restrict__ ao) {
  __shared__ __align__(16) unsigned short kt_lds[32 * 64];   // [key][feat]
  __shared__ __align__(16) unsigned short vt_lds[64 * 32];   // [feat][key]
  __shared__ __align__(16) unsigned short p_lds[4][16 * 32];
  int tid = threadIdx.x, lane = tid & 31, w = tid >> 5;
  int blk = blockIdx.x;
  int tile = blk & 31, bh = blk >> 5;
  int b = bh >> 4, h = bh & 15;
  int i0 = tile * 64 + w * 16;
  int hi = lane >> 4, ln = lane & 15;

  const unsigned short* qrow = qe + ((size_t)bh * kN + i0) * 64;
  v16bf aq[2];
  aq[0] = load_a_frag(qrow, 64, lane);
  aq[1] = load_a_frag(qrow + 32, 64, lane);

  v8f accO[4];
#pragma unroll
  for (int g = 0; g < 4; ++g) accO[g] = v8f_zero();
  float mrow[8], lrow[8];
#pragma unroll
  for (int r = 0; r < 8; ++r) { mrow[r] = -1e30f; lrow[r] = 0.f; }

  for (int kb = 0; kb < kN; kb += 32) {
    // stage keff chunk [key][feat] (async when available)
#pragma unroll
    for (int q = 0; q < 2; ++q) {
      int idx = tid * 16 + q * 8;
      int r = idx >> 6, c = idx & 63;
      copy16_g2l(&ke[((size_t)bh * kN + kb + r) * 64 + c], &kt_lds[r * 64 + c]);
    }
    // stage v chunk transposed [feat][key]
    {
      int kk = tid >> 2;              // 0..31 key
      int f0 = (tid & 3) * 16;        // feature base
      const unsigned short* gp = &vb[((size_t)bh * kN + kb + kk) * 64 + f0];
      uint4 t0 = *(const uint4*)gp;
      uint4 t1 = *(const uint4*)(gp + 8);
      const unsigned short* s0 = (const unsigned short*)&t0;
      const unsigned short* s1 = (const unsigned short*)&t1;
#pragma unroll
      for (int q = 0; q < 8; ++q) {
        vt_lds[(f0 + q) * 32 + kk]     = s0[q];
        vt_lds[(f0 + 8 + q) * 32 + kk] = s1[q];
      }
    }
    async_wait();
    __syncthreads();

    // S = Q @ K^T for 32 keys (two 16x16 tiles)
    v8f accS[2];
    accS[0] = v8f_zero(); accS[1] = v8f_zero();
#pragma unroll
    for (int jt = 0; jt < 2; ++jt)
#pragma unroll
      for (int kf = 0; kf < 2; ++kf)
        accS[jt] = wmma_bf16(aq[kf], load_b_frag(&kt_lds[(jt * 16) * 64 + kf * 32], 64, lane), accS[jt]);

    // online softmax (row lives across 16 lanes of one half-wave)
#pragma unroll
    for (int r = 0; r < 8; ++r) {
      float vmax = fmaxf(accS[0][r], accS[1][r]);
#pragma unroll
      for (int off = 8; off >= 1; off >>= 1) vmax = fmaxf(vmax, __shfl_xor(vmax, off, 32));
      float mnew = fmaxf(mrow[r], vmax);
      float alpha = __expf(mrow[r] - mnew);
      float p0 = __expf(accS[0][r] - mnew);
      float p1 = __expf(accS[1][r] - mnew);
      float rsum = p0 + p1;
#pragma unroll
      for (int off = 8; off >= 1; off >>= 1) rsum += __shfl_xor(rsum, off, 32);
      lrow[r] = lrow[r] * alpha + rsum;
      mrow[r] = mnew;
#pragma unroll
      for (int g = 0; g < 4; ++g) accO[g][r] = accO[g][r] * alpha;
      p_lds[w][(r + 8 * hi) * 32 + ln]      = f32_to_bf16(p0);
      p_lds[w][(r + 8 * hi) * 32 + 16 + ln] = f32_to_bf16(p1);
    }
    __syncthreads();

    // O += P @ V
    v16bf ap = load_a_frag(&p_lds[w][0], 32, lane);
#pragma unroll
    for (int g = 0; g < 4; ++g)
      accO[g] = wmma_bf16(ap, load_b_frag(&vt_lds[(g * 16) * 32], 32, lane), accO[g]);
    __syncthreads();
  }

  // normalize + store bf16 into (B,N,DM)
  unsigned short* op = ao + ((size_t)b * kN + i0) * kDM + h * 64;
#pragma unroll
  for (int g = 0; g < 4; ++g)
#pragma unroll
    for (int r = 0; r < 8; ++r) {
      float o = accO[g][r] / lrow[r];
      op[(r + 8 * hi) * kDM + g * 16 + ln] = f32_to_bf16(o);
    }
}

// ---------------------------------------------------------------------------
extern "C" void kernel_launch(void* const* d_in, const int* in_sizes, int n_in,
                              void* d_out, int out_size, void* d_ws, size_t ws_size,
                              hipStream_t stream) {
  const float* x     = (const float*)d_in[0];
  const float* w_qkv = (const float*)d_in[1];
  const float* b_qkv = (const float*)d_in[2];
  const float* w_out = (const float*)d_in[3];
  const float* b_out = (const float*)d_in[4];
  const float* emb   = (const float*)d_in[5];
  const int*   lidx  = (const int*)d_in[6];

  char* ws = (char*)d_ws;
  size_t off = 0;
  auto take = [&](size_t bytes) -> char* {
    char* p = ws + off;
    off = (off + bytes + 255) & ~(size_t)255;
    return p;
  };
  unsigned short* x_bf    = (unsigned short*)take((size_t)kB * kN * kDM * 2);       // 8 MB
  unsigned short* wqkv_t  = (unsigned short*)take((size_t)kDM * 3 * kDM * 2);       // 6 MB  [3072][1024]
  unsigned short* wout_t  = (unsigned short*)take((size_t)kDM * kDM * 2);           // 2 MB  [1024][1024]
  float*          qkv_f   = (float*)take((size_t)kB * kN * 3 * kDM * 4);            // 48 MB
  unsigned short* q_bf    = (unsigned short*)take((size_t)kBH * kN * kHD * 2);      // 8 MB
  unsigned short* k_bf    = (unsigned short*)take((size_t)kBH * kN * kHD * 2);      // 8 MB
  unsigned short* v_bf    = (unsigned short*)take((size_t)kBH * kN * kHD * 2);      // 8 MB
  unsigned short* kl_bf   = (unsigned short*)take((size_t)kBH * kM * kHD * 2);      // 256 KB
  float*          k2inv   = (float*)take((size_t)kBH * kM * kM * 4);                // 512 KB
  float*          rb      = (float*)take((size_t)kBH * kN * kM * 4);                // 16 MB
  unsigned short* qeff    = (unsigned short*)take((size_t)kBH * kN * kM * 2);       // 8 MB
  unsigned short* keff    = (unsigned short*)take((size_t)kBH * kN * kM * 2);       // 8 MB
  unsigned short* attn_bf = (unsigned short*)take((size_t)kB * kN * kDM * 2);       // 8 MB

  // 1) activations -> bf16; weights -> transposed bf16 ([n][k])
  cvt_kernel<<<(kB * kN * kDM / 4 + 255) / 256, 256, 0, stream>>>(x, x_bf, kB * kN * kDM / 4);
  {
    dim3 g(3 * kDM / 32, kDM / 32);
    transpose_cvt<<<g, 256, 0, stream>>>(w_qkv, wqkv_t, kDM, 3 * kDM);
  }
  {
    dim3 g(kDM / 32, kDM / 32);
    transpose_cvt<<<g, 256, 0, stream>>>(w_out, wout_t, kDM, kDM);
  }

  // 2) QKV projection (4096x1024 @ 1024x3072)
  {
    dim3 g(3 * kDM / 128, kB * kN / 128);
    gemm_bf16<<<g, 256, 0, stream>>>(x_bf, wqkv_t, b_qkv, qkv_f, kB * kN, 3 * kDM, kDM);
  }

  // 3) repack to (B,H,N,HD) bf16
  repack_qkv<<<(kB * kN * 3 * kDM + 255) / 256, 256, 0, stream>>>(qkv_f, q_bf, k_bf, v_bf);

  // 4) landmark gather
  gather_kl<<<(kBH * kM * kHD + 255) / 256, 256, 0, stream>>>(k_bf, lidx, kl_bf);

  // 5) relative position bias
  rel_bias_kernel<<<kBH * kN / 4, 256, 0, stream>>>(q_bf, emb, rb);

  // 6) pinv(kernel2) via Newton-Schulz
  pinv_kernel<<<kBH, 256, 0, stream>>>(kl_bf, k2inv);

  // 7) effective queries/keys
  form_eff<<<kBH * (kN / 64), 128, 0, stream>>>(q_bf, kl_bf, rb, k2inv, qeff, 1);
  form_eff<<<kBH * (kN / 64), 128, 0, stream>>>(k_bf, kl_bf, rb, k2inv, keff, 0);

  // 8) streaming softmax attention
  flash_kernel<<<kBH * (kN / 64), 128, 0, stream>>>(qeff, keff, v_bf, attn_bf);

  // 9) output projection (4096x1024 @ 1024x1024) -> f32 d_out
  {
    dim3 g(kDM / 128, kB * kN / 128);
    gemm_bf16<<<g, 256, 0, stream>>>(attn_bf, wout_t, b_out, (float*)d_out, kB * kN, kDM, kDM);
  }
  (void)in_sizes; (void)n_in; (void)out_size; (void)ws_size;
}
